// StreamingMultiheadAttention_9423158247867
// MI455X (gfx1250) — matrix operand order repaired and checked by reading, not compiled, not run on hardware
//
#include <hip/hip_runtime.h>
#include <hip/hip_fp16.h>
#include <cstdint>

typedef __attribute__((ext_vector_type(16))) _Float16     v16h;
typedef __attribute__((ext_vector_type(8)))  float        v8f;
typedef __attribute__((ext_vector_type(4)))  unsigned int u32x4;
typedef __attribute__((ext_vector_type(8)))  unsigned int u32x8;

#define E_DIM   1024
#define N_HEADS 16
#define H_DIM   64
#define SEQ     2048
#define BATCH   2
#define ROWS    (BATCH * SEQ)   // 4096
#define QKV_N   (3 * E_DIM)     // 3072

// ---------------------------------------------------------------------------
// XOR lane shuffle via ds_swizzle_b32 (group-of-32 mode: xor<<10 | and 0x1f).
// Masks 1/2/4/8 stay within each 16-lane half — matching the WMMA C/D layout
// where row m lives on lanes 0-15 (M=r) / 16-31 (M=r+8).
// ---------------------------------------------------------------------------
template <int MASK>
__device__ __forceinline__ float xor_swz(float x) {
    const int r = __builtin_amdgcn_ds_swizzle(__float_as_int(x), (MASK << 10) | 0x1f);
    return __int_as_float(r);
}
__device__ __forceinline__ float red_max16(float x) {
    x = fmaxf(x, xor_swz<1>(x));
    x = fmaxf(x, xor_swz<2>(x));
    x = fmaxf(x, xor_swz<4>(x));
    x = fmaxf(x, xor_swz<8>(x));
    return x;
}
__device__ __forceinline__ float red_sum16(float x) {
    x += xor_swz<1>(x);
    x += xor_swz<2>(x);
    x += xor_swz<4>(x);
    x += xor_swz<8>(x);
    return x;
}

// ---------------------------------------------------------------------------
// LDS fragment loaders for V_WMMA_F32_16X16X32_F16 (contiguous k order, A and B alike).
// Tiles are row-major [rows x stride] halves; stride*2 bytes is a multiple of
// 16 for ds_load_b128 (40 -> 80B, 72 -> 144B), both conflict-free bank maps.
//
// A (16x32, MxK): lane<16 -> M=lane, K=[0..15];
//                 lane>=16 -> M=lane-16, K=[16..31].
// B (32x16, KxN): lane<16 -> N=lane, K=[0..15]; lane>=16 -> N=lane-16, K=[16..31].
// ---------------------------------------------------------------------------
__device__ __forceinline__ v16h lds_frag_a(const _Float16* tile, int stride,
                                           int rowbase, int koff, int lane) {
    const _Float16* p = tile + (rowbase + (lane & 15)) * stride + koff + ((lane >> 4) << 4);
    union { v16h h; uint4 u[2]; } f;
    f.u[0] = *(const uint4*)(p);
    f.u[1] = *(const uint4*)(p + 8);
    return f.h;
}
__device__ __forceinline__ v16h lds_frag_b(const _Float16* tile, int stride,
                                           int rowbase, int koff, int lane) {
    const _Float16* p = tile + (rowbase + (lane & 15)) * stride + koff + ((lane >> 4) << 4);
    union { v16h h; uint4 u[2]; } f;
    f.u[0] = *(const uint4*)(p);
    f.u[1] = *(const uint4*)(p + 8);
    return f.h;
}

// ---------------------------------------------------------------------------
// Tensor Data Mover: DMA a 64x64 f16 tile (row stride QKV_N halves) from
// global into a padded LDS tile (stride 72 halves = 128B data + 16B pad).
// D# per CDNA5 ISA 8.3/8.4: group0 = {count|lds_addr|global_addr|type=2},
// group1 = {data_size=2B, pad_enable, pad_interval=32dw(code 4),
//           pad_amount=4dw(code 3), dims/strides}.
// Issued by one wave; completion via TENSORcnt (in-order per wave).
// ---------------------------------------------------------------------------
__device__ __forceinline__ void tdm_load_tile64(unsigned lds_byte_off,
                                                const _Float16* gsrc) {
    const uint64_t ga = (uint64_t)(uintptr_t)gsrc;
    u32x4 g0;
    g0[0] = 1u;                                   // count=1, user descriptor
    g0[1] = lds_byte_off;                         // lds_addr
    g0[2] = (unsigned)ga;                         // global_addr[31:0]
    g0[3] = (unsigned)(ga >> 32) | 0x80000000u;   // global_addr[56:32] | type=2
    u32x8 g1;
    g1[0] = (1u << 16)      // data_size = 2 bytes
          | (1u << 20)      // pad_enable
          | (4u << 22)      // pad_interval: 32 DWORDs (=128B row)
          | (3u << 25);     // pad_amount:   4 DWORDs (=16B)
    g1[1] = (unsigned)H_DIM << 16;   // tensor_dim0 = 64 (elements, contiguous d)
    g1[2] = (unsigned)SEQ << 16;     // tensor_dim1 = 2048 rows
    g1[3] = (unsigned)H_DIM << 16;   // tile_dim0 = 64
    g1[4] = 64u;                     // tile_dim1 = 64 rows
    g1[5] = (unsigned)QKV_N;         // tensor_dim0_stride = 3072 elements
    g1[6] = 0u;
    g1[7] = 0u;
    asm volatile("tensor_load_to_lds %0, %1" :: "s"(g0), "s"(g1) : "memory");
}

// ---------------------------------------------------------------------------
// C = A[MxK] * W[NxK]^T + bias, f16 WMMA with f32 accumulate.
// 256 threads (8 waves), 128x128 C tile, waves 4(M) x 2(N), 32x64 per wave.
// global_prefetch of the next K-step tiles overlaps the current WMMAs.
// ---------------------------------------------------------------------------
template <typename AT, typename OT>
__global__ __launch_bounds__(256)
void gemm_bias_wmma(const AT* __restrict__ A, const float* __restrict__ W,
                    const float* __restrict__ bias, OT* __restrict__ C,
                    int M, int N, int K) {
    constexpr int KS = 40;                  // padded LDS stride (halves)
    __shared__ _Float16 As[128 * KS];
    __shared__ _Float16 Bs[128 * KS];

    const int tid  = threadIdx.x;
    const int lane = tid & 31;
    const int wave = tid >> 5;
    const int wm   = wave >> 1;             // 0..3
    const int wn   = wave & 1;              // 0..1
    const int m0   = blockIdx.y * 128;
    const int n0   = blockIdx.x * 128;

    const v8f vzero = {0.f, 0.f, 0.f, 0.f, 0.f, 0.f, 0.f, 0.f};
    v8f acc[2][4];
#pragma unroll
    for (int i = 0; i < 2; ++i)
#pragma unroll
        for (int j = 0; j < 4; ++j) acc[i][j] = vzero;

    for (int kt = 0; kt < K; kt += 32) {
        // ---- stage A tile (convert f32->f16 in flight if needed) ----
#pragma unroll
        for (int i = 0; i < 4; ++i) {
            const int s   = tid + i * 256;          // 1024 chunks of 4 elems
            const int row = s >> 3;
            const int kv  = (s & 7) * 4;
            union { uint2 u; _Float16 h4[4]; } pk;
            if constexpr (sizeof(AT) == 4) {
                const float4 v = *(const float4*)(A + (size_t)(m0 + row) * K + kt + kv);
                pk.h4[0] = (_Float16)v.x; pk.h4[1] = (_Float16)v.y;
                pk.h4[2] = (_Float16)v.z; pk.h4[3] = (_Float16)v.w;
            } else {
                pk.u = *(const uint2*)(A + (size_t)(m0 + row) * K + kt + kv);
            }
            *(uint2*)&As[row * KS + kv] = pk.u;
            if (i == 0 && kt + 32 < K)   // prefetch next K-step (global_prefetch_b8)
                __builtin_prefetch(A + (size_t)(m0 + row) * K + kt + 32 + kv, 0, 3);
        }
        // ---- stage W tile (always f32 -> f16) ----
#pragma unroll
        for (int i = 0; i < 4; ++i) {
            const int s   = tid + i * 256;
            const int row = s >> 3;
            const int kv  = (s & 7) * 4;
            const float4 v = *(const float4*)(W + (size_t)(n0 + row) * K + kt + kv);
            union { uint2 u; _Float16 h4[4]; } pk;
            pk.h4[0] = (_Float16)v.x; pk.h4[1] = (_Float16)v.y;
            pk.h4[2] = (_Float16)v.z; pk.h4[3] = (_Float16)v.w;
            *(uint2*)&Bs[row * KS + kv] = pk.u;
            if (i == 0 && kt + 32 < K)
                __builtin_prefetch(W + (size_t)(n0 + row) * K + kt + 32 + kv, 0, 3);
        }
        __syncthreads();

        v16h af[2], bf[4];
#pragma unroll
        for (int i = 0; i < 2; ++i) af[i] = lds_frag_a(As, KS, wm * 32 + i * 16, 0, lane);
#pragma unroll
        for (int j = 0; j < 4; ++j) bf[j] = lds_frag_b(Bs, KS, wn * 64 + j * 16, 0, lane);
#pragma unroll
        for (int i = 0; i < 2; ++i)
#pragma unroll
            for (int j = 0; j < 4; ++j)
                acc[i][j] = __builtin_amdgcn_wmma_f32_16x16x32_f16(
                    false, af[i], false, bf[j], (short)0, acc[i][j], false, false);
        __syncthreads();
    }

    // ---- epilogue: C/D layout -> lane<16: M=r, lane>=16: M=r+8; N=lane&15 ----
    const int rL = (lane >> 4) << 3;
    const int cL = lane & 15;
#pragma unroll
    for (int i = 0; i < 2; ++i) {
#pragma unroll
        for (int j = 0; j < 4; ++j) {
            const int col = n0 + wn * 64 + j * 16 + cL;
            const float bv = bias[col];
#pragma unroll
            for (int r = 0; r < 8; ++r) {
                const int rowm = m0 + wm * 32 + i * 16 + rL + r;
                C[(size_t)rowm * N + col] = (OT)(acc[i][j][r] + bv);
            }
        }
    }
}

// ---------------------------------------------------------------------------
// Flash attention: grid = (T/128, B*H), 256 threads, 16 query rows per wave.
// K blocks are staged by the Tensor Data Mover, double-buffered: wave 0
// issues the TDM for block kb+1, then waits TENSORcnt<=1 (in-order) so the
// DMA overlaps this block's WMMA + softmax. V is staged transposed manually.
// ---------------------------------------------------------------------------
__global__ __launch_bounds__(256)
void flash_attn_wmma(const _Float16* __restrict__ qkv, _Float16* __restrict__ out) {
    constexpr int QS = 72;                   // padded LDS stride (halves)
    constexpr int NB = SEQ / 64;             // 32 key blocks
    __shared__ _Float16 Qs[128 * QS];
    __shared__ _Float16 Ks[2][64 * QS];      // TDM double buffer
    __shared__ _Float16 Vt[64 * QS];         // transposed: [d][n]
    __shared__ _Float16 Ps[128 * QS];        // 8 waves x 16 rows

    const int tid  = threadIdx.x;
    const int lane = tid & 31;
    const int wave = tid >> 5;
    const int bh   = blockIdx.y;
    const int b    = bh >> 4;
    const int h    = bh & 15;
    const int t0   = blockIdx.x * 128;

    const size_t rowstride = QKV_N;  // 3072 halves per (b,t) row
    const size_t baseQ = (size_t)(b * SEQ) * rowstride + (size_t)h * H_DIM;
    const size_t baseK = baseQ + E_DIM;
    const size_t baseV = baseQ + 2 * E_DIM;

    // ---- kick off TDM for K block 0 ----
    if (wave == 0)
        tdm_load_tile64((unsigned)(uintptr_t)&Ks[0][0], qkv + baseK);

    // ---- stage Q once, pre-scaled by D^-0.5 = 0.125 (exact in f16) ----
#pragma unroll
    for (int i = 0; i < 8; ++i) {
        const int s   = tid + i * 256;       // 2048 chunks of 4 halves
        const int row = s >> 4;
        const int d   = (s & 15) * 4;
        union { uint2 u; _Float16 h4[4]; } pk;
        pk.u = *(const uint2*)(qkv + baseQ + (size_t)(t0 + row) * rowstride + d);
#pragma unroll
        for (int c = 0; c < 4; ++c) pk.h4[c] = pk.h4[c] * (_Float16)0.125f;
        *(uint2*)&Qs[row * QS + d] = pk.u;
    }

    const v8f vzero = {0.f, 0.f, 0.f, 0.f, 0.f, 0.f, 0.f, 0.f};
    v8f o[4];
#pragma unroll
    for (int j = 0; j < 4; ++j) o[j] = vzero;
    float mi[8], li[8];
#pragma unroll
    for (int r = 0; r < 8; ++r) { mi[r] = -3.0e38f; li[r] = 0.f; }

    __syncthreads();

    for (int kb = 0; kb < NB; ++kb) {
        const int n0 = kb * 64;
        // ---- wave 0: issue TDM for the NEXT K block (other buffer) ----
        if (wave == 0 && kb + 1 < NB)
            tdm_load_tile64((unsigned)(uintptr_t)&Ks[(kb + 1) & 1][0],
                            qkv + baseK + (size_t)(n0 + 64) * rowstride);
        // ---- stage V block transposed: Vt[d][n] ----
#pragma unroll
        for (int i = 0; i < 4; ++i) {
            const int s   = tid + i * 256;
            const int row = s >> 4;
            const int d   = (s & 15) * 4;
            union { uint2 u; _Float16 h4[4]; } pk;
            pk.u = *(const uint2*)(qkv + baseV + (size_t)(n0 + row) * rowstride + d);
#pragma unroll
            for (int c = 0; c < 4; ++c) Vt[(d + c) * QS + row] = pk.h4[c];
        }
        // ---- wave 0: ensure TDM for THIS block finished (in-order cnt) ----
        if (wave == 0) {
            if (kb + 1 < NB) __builtin_amdgcn_s_wait_tensorcnt(1);
            else             __builtin_amdgcn_s_wait_tensorcnt(0);
        }
        __syncthreads();

        // ---- S = Q * K^T  (16 x 64 per wave) ----
        const _Float16* Kcur = Ks[kb & 1];
        v8f s4[4];
#pragma unroll
        for (int j = 0; j < 4; ++j) s4[j] = vzero;
#pragma unroll
        for (int kk = 0; kk < H_DIM; kk += 32) {
            const v16h aq = lds_frag_a(Qs, QS, wave * 16, kk, lane);
#pragma unroll
            for (int j = 0; j < 4; ++j) {
                const v16h bk = lds_frag_b(Kcur, QS, j * 16, kk, lane);
                s4[j] = __builtin_amdgcn_wmma_f32_16x16x32_f16(
                    false, aq, false, bk, (short)0, s4[j], false, false);
            }
        }

        // ---- online softmax; row m=r(+8) lives on lanes 0-15 / 16-31 ----
        float pv[4][8];
#pragma unroll
        for (int r = 0; r < 8; ++r) {
            float mx = fmaxf(fmaxf(s4[0][r], s4[1][r]), fmaxf(s4[2][r], s4[3][r]));
            mx = red_max16(mx);
            const float mnew = fmaxf(mi[r], mx);
            const float corr = __expf(mi[r] - mnew);
            float rs = 0.f;
#pragma unroll
            for (int j = 0; j < 4; ++j) {
                const float p = __expf(s4[j][r] - mnew);
                pv[j][r] = p;
                rs += p;
            }
            rs = red_sum16(rs);
            li[r] = li[r] * corr + rs;
            mi[r] = mnew;
#pragma unroll
            for (int j = 0; j < 4; ++j) o[j][r] = o[j][r] * corr;
        }

        // ---- write P (f16) to per-wave LDS region for A-fragment re-layout ----
        _Float16* Pw = Ps + wave * 16 * QS;
        {
            const int rL = (lane >> 4) << 3;
            const int cL = lane & 15;
#pragma unroll
            for (int j = 0; j < 4; ++j)
#pragma unroll
                for (int r = 0; r < 8; ++r)
                    Pw[(rL + r) * QS + j * 16 + cL] = (_Float16)pv[j][r];
        }
        __syncthreads();

        // ---- O += P * V ----
#pragma unroll
        for (int kk = 0; kk < 64; kk += 32) {
            const v16h ap = lds_frag_a(Pw, QS, 0, kk, lane);
#pragma unroll
            for (int j = 0; j < 4; ++j) {
                const v16h bv = lds_frag_b(Vt, QS, j * 16, kk, lane);
                o[j] = __builtin_amdgcn_wmma_f32_16x16x32_f16(
                    false, ap, false, bv, (short)0, o[j], false, false);
            }
        }
        __syncthreads();
    }

    // ---- normalize and store attention output (f16, [B*T, E]) ----
    const int rL = (lane >> 4) << 3;
    const int cL = lane & 15;
#pragma unroll
    for (int r = 0; r < 8; ++r) {
        const float inv = 1.f / li[r];
        const int trow = t0 + wave * 16 + rL + r;
        const size_t orow = (size_t)(b * SEQ + trow) * E_DIM + (size_t)h * H_DIM;
#pragma unroll
        for (int j = 0; j < 4; ++j)
            out[orow + j * 16 + cL] = (_Float16)(o[j][r] * inv);
    }
}

// ---------------------------------------------------------------------------
extern "C" void kernel_launch(void* const* d_in, const int* in_sizes, int n_in,
                              void* d_out, int out_size, void* d_ws, size_t ws_size,
                              hipStream_t stream) {
    (void)in_sizes; (void)n_in; (void)out_size; (void)ws_size;
    const float* query = (const float*)d_in[0];
    const float* Wqkv  = (const float*)d_in[1];
    const float* bqkv  = (const float*)d_in[2];
    const float* Wo    = (const float*)d_in[3];
    const float* bo    = (const float*)d_in[4];
    float* out = (float*)d_out;

    _Float16* qkv_h  = (_Float16*)d_ws;                       // [4096 x 3072] f16
    _Float16* attn_h = qkv_h + (size_t)ROWS * QKV_N;          // [4096 x 1024] f16

    // 1) fused QKV projection (f32 in, f16 out)
    gemm_bias_wmma<float, _Float16>
        <<<dim3(QKV_N / 128, ROWS / 128), 256, 0, stream>>>(
            query, Wqkv, bqkv, qkv_h, ROWS, QKV_N, E_DIM);

    // 2) flash attention over heads (TDM-staged K blocks)
    flash_attn_wmma<<<dim3(SEQ / 128, BATCH * N_HEADS), 256, 0, stream>>>(qkv_h, attn_h);

    // 3) output projection (f16 in, f32 out)
    gemm_bias_wmma<_Float16, float>
        <<<dim3(E_DIM / 128, ROWS / 128), 256, 0, stream>>>(
            attn_h, Wo, bo, out, ROWS, E_DIM, E_DIM);
}
